// DIFSRAttention_45414984187920
// MI455X (gfx1250) — matrix operand order repaired and checked by reading, not compiled, not run on hardware
//
#include <hip/hip_runtime.h>
#include <hip/hip_bf16.h>

// ---------------------------------------------------------------------------
// DIF-SR attention for MI455X (gfx1250, wave32, WMMA).
// B=16, L=512, HID=1024, NH=16, HD=64.
// All matmuls use v_wmma_f32_16x16x32_bf16 (f32 accumulate).
// Workspace layout (bf16, each tensor 8192x1024 = 16 MiB): Qid,Kid,Qc,Kc,Qp,Kp,V,AttnOut.
// ---------------------------------------------------------------------------

typedef __attribute__((ext_vector_type(16))) __bf16 v16bf;
typedef __attribute__((ext_vector_type(8)))  __bf16 v8bf;
typedef __attribute__((ext_vector_type(4)))  __bf16 v4bf;
typedef __attribute__((ext_vector_type(8)))  float  v8f;

#define HIDDIM 1024
#define SEQ    512
#define NHEAD  16
#define HEADD  64
#define NEG_INF (-__builtin_inff())

static __device__ __forceinline__ v8f wmma_bf16(v16bf a, v16bf b, v8f c) {
  // (neg_a, A, neg_b, B, c_mod, C, reuse_a, reuse_b)
  return __builtin_amdgcn_wmma_f32_16x16x32_bf16(false, a, false, b, (short)0, c, false, false);
}

static __device__ __forceinline__ v16bf pack16(v8bf lo, v8bf hi8) {
  v16bf r;
#pragma unroll
  for (int i = 0; i < 8; ++i) { r[i] = lo[i]; r[i + 8] = hi8[i]; }
  return r;
}

static __device__ __forceinline__ float redmax16(float v) {
#pragma unroll
  for (int m = 1; m < 16; m <<= 1) v = fmaxf(v, __shfl_xor(v, m, 32));
  return v;
}
static __device__ __forceinline__ float redsum16(float v) {
#pragma unroll
  for (int m = 1; m < 16; m <<= 1) v += __shfl_xor(v, m, 32);
  return v;
}

// ---------------------------------------------------------------------------
// GEMM: Out[8192,1024] = A[8192,1024] @ W[1024,1024] + bias
// A is f32 (converted to bf16 on the LDS stage) or bf16; Out is bf16 or f32.
// Block 256 threads = 8 waves; block tile 128x128; wave tile 32x64.
// ---------------------------------------------------------------------------
template <bool A_BF16, bool OUT_F32>
__global__ __launch_bounds__(256) void gemm1024(const void* __restrict__ Ap,
                                                const float* __restrict__ W,
                                                const float* __restrict__ bias,
                                                void* __restrict__ Outp) {
  // stride 40 halves = 80B: 16B-aligned b128 frags, conflict-free (20*r mod 64 distinct)
  __shared__ alignas(16) __bf16 sA[128 * 40];
  __shared__ alignas(16) __bf16 sB[128 * 40];  // [n][k]

  const int tid  = threadIdx.x;
  const int lane = tid & 31, wid = tid >> 5;
  const int hi   = lane >> 4, ln = lane & 15;
  const int m0 = blockIdx.y * 128, n0 = blockIdx.x * 128;
  const int mb = (wid >> 1) * 32, nb = (wid & 1) * 64;

  v8f acc[2][4];
#pragma unroll
  for (int a = 0; a < 2; ++a)
#pragma unroll
    for (int c = 0; c < 4; ++c)
#pragma unroll
      for (int i = 0; i < 8; ++i) acc[a][c][i] = 0.0f;

  for (int kk = 0; kk < 32; ++kk) {
    const int k0 = kk * 32;
    if (A_BF16) {
      const __bf16* A = (const __bf16*)Ap;
      for (int c = tid; c < 512; c += 256) {
        int row = c >> 2, ch = c & 3;
        v8bf v = *(const v8bf*)(A + (size_t)(m0 + row) * HIDDIM + k0 + ch * 8);
        *(v8bf*)(sA + row * 40 + ch * 8) = v;
      }
    } else {
      const float* A = (const float*)Ap;
      for (int c = tid; c < 1024; c += 256) {
        int row = c >> 3, ch = c & 7;
        float4 v = *(const float4*)(A + (size_t)(m0 + row) * HIDDIM + k0 + ch * 4);
        v4bf bv;
        bv[0] = (__bf16)v.x; bv[1] = (__bf16)v.y; bv[2] = (__bf16)v.z; bv[3] = (__bf16)v.w;
        *(v4bf*)(sA + row * 40 + ch * 4) = bv;
      }
    }
    // W[k][n] f32 -> sB[n][k] bf16 (transposed for B-fragments)
    for (int c = tid; c < 1024; c += 256) {
      int kr = c >> 5, nch = c & 31;
      float4 v = *(const float4*)(W + (size_t)(k0 + kr) * HIDDIM + n0 + nch * 4);
      sB[(nch * 4 + 0) * 40 + kr] = (__bf16)v.x;
      sB[(nch * 4 + 1) * 40 + kr] = (__bf16)v.y;
      sB[(nch * 4 + 2) * 40 + kr] = (__bf16)v.z;
      sB[(nch * 4 + 3) * 40 + kr] = (__bf16)v.w;
    }
    __syncthreads();

    v16bf bf[4];
#pragma unroll
    for (int ns = 0; ns < 4; ++ns) {
      const __bf16* p = sB + (nb + ns * 16 + ln) * 40 + hi * 16;
      bf[ns] = pack16(*(const v8bf*)p, *(const v8bf*)(p + 8));
    }
#pragma unroll
    for (int ms = 0; ms < 2; ++ms) {
      const __bf16* p = sA + (mb + ms * 16 + ln) * 40 + hi * 16;
      v16bf af = pack16(*(const v8bf*)p, *(const v8bf*)(p + 8));
#pragma unroll
      for (int ns = 0; ns < 4; ++ns) acc[ms][ns] = wmma_bf16(af, bf[ns], acc[ms][ns]);
    }
    __syncthreads();
  }

#pragma unroll
  for (int ns = 0; ns < 4; ++ns) {
    const int n  = n0 + nb + ns * 16 + ln;
    const float bv = bias[n];
#pragma unroll
    for (int ms = 0; ms < 2; ++ms) {
#pragma unroll
      for (int i = 0; i < 8; ++i) {
        const int   row = m0 + mb + ms * 16 + hi * 8 + i;
        const float val = acc[ms][ns][i] + bv;
        if (OUT_F32) ((float*)Outp)[(size_t)row * HIDDIM + n] = val;
        else         ((__bf16*)Outp)[(size_t)row * HIDDIM + n] = (__bf16)val;
      }
    }
  }
}

// ---------------------------------------------------------------------------
// Flash-style causal attention with 3 summed QK^T scores + relative_time bias.
// Grid: B*NH*(L/128) blocks of 256 threads (8 waves, 16 queries per wave).
// Wave tile kept at 16 queries so live VGPRs stay well under 256 (no
// s_set_vgpr_msb thrash, higher occupancy). relative_time streamed once,
// with one-block-ahead global_prefetch_b8.
// ---------------------------------------------------------------------------
__global__ __launch_bounds__(256) void attn_kernel(
    const __bf16* __restrict__ Qid, const __bf16* __restrict__ Kid,
    const __bf16* __restrict__ Qc,  const __bf16* __restrict__ Kc,
    const __bf16* __restrict__ Qp,  const __bf16* __restrict__ Kp,
    const __bf16* __restrict__ Vb,  const float* __restrict__ rel,
    __bf16* __restrict__ Ob) {
  // stride 72 halves = 144B: 16B aligned, conflict-free (36*r mod 64 distinct)
  __shared__ alignas(16) __bf16 sK[3][64 * 72];   // [mat][key][d]
  __shared__ alignas(16) __bf16 sVt[64 * 72];     // [d][key]
  __shared__ alignas(16) __bf16 sP[8][16 * 72];   // per-wave P tile [q][key]

  const int tid = threadIdx.x, lane = tid & 31, wid = tid >> 5;
  const int hi = lane >> 4, ln = lane & 15;
  const int blk = blockIdx.x;
  const int qc = blk & 3, bh = blk >> 2;
  const int b = bh >> 4, h = bh & 15;
  const int row0 = b * SEQ;
  const int col0 = h * HEADD;
  const int qwg = qc * 128;
  const int qw  = qwg + wid * 16;   // this wave's 16 queries

  const __bf16* Qm[3] = {Qid, Qc, Qp};
  const __bf16* Km[3] = {Kid, Kc, Kp};
  const float* relp = rel + (size_t)bh * SEQ * SEQ;

  v8f   o[4];
  float mrow[8], lrow[8];
#pragma unroll
  for (int i = 0; i < 8; ++i) { mrow[i] = NEG_INF; lrow[i] = 0.0f; }
#pragma unroll
  for (int nd = 0; nd < 4; ++nd)
#pragma unroll
    for (int i = 0; i < 8; ++i) o[nd][i] = 0.0f;

  const int nkb = qwg / 64 + 2;  // causal: only key blocks with k0 <= max q in WG
  for (int kb = 0; kb < nkb; ++kb) {
    const int k0 = kb * 64;
    // ---- stage K tiles (3 mats) and transposed V tile ----
    for (int mat = 0; mat < 3; ++mat)
      for (int c = tid; c < 512; c += 256) {
        int r = c >> 3, ch = c & 7;
        v8bf v = *(const v8bf*)(Km[mat] + (size_t)(row0 + k0 + r) * HIDDIM + col0 + ch * 8);
        *(v8bf*)(&sK[mat][r * 72 + ch * 8]) = v;
      }
    for (int c = tid; c < 512; c += 256) {
      int r = c >> 3, ch = c & 7;
      v8bf v = *(const v8bf*)(Vb + (size_t)(row0 + k0 + r) * HIDDIM + col0 + ch * 8);
#pragma unroll
      for (int j = 0; j < 8; ++j) sVt[(ch * 8 + j) * 72 + r] = v[j];
    }
    __syncthreads();

    if (k0 <= qw + 15) {  // wave has unmasked work in this key block
      // prefetch next key block's relative_time rows (dominant HBM stream)
      if (kb + 1 < nkb) {
#pragma unroll
        for (int i = 0; i < 8; ++i) {
          const int q = qw + hi * 8 + i;
          __builtin_prefetch(&relp[(size_t)q * SEQ + k0 + 64 + ln * 4], 0, 0);
        }
      }

      v8f s[4];
#pragma unroll
      for (int ns = 0; ns < 4; ++ns)
#pragma unroll
        for (int i = 0; i < 8; ++i) s[ns][i] = 0.0f;

      // ---- S = sum of 3 Q K^T ----
#pragma unroll
      for (int mat = 0; mat < 3; ++mat) {
#pragma unroll
        for (int ks = 0; ks < 2; ++ks) {
          const int kbase = ks * 32;
          v16bf bf[4];
#pragma unroll
          for (int ns = 0; ns < 4; ++ns) {
            const __bf16* p = &sK[mat][(ns * 16 + ln) * 72 + kbase + hi * 16];
            bf[ns] = pack16(*(const v8bf*)p, *(const v8bf*)(p + 8));
          }
          const __bf16* g = Qm[mat] + (size_t)(row0 + qw + ln) * HIDDIM +
                            col0 + kbase + hi * 16;
          v16bf af = pack16(*(const v8bf*)g, *(const v8bf*)(g + 8));
#pragma unroll
          for (int ns = 0; ns < 4; ++ns) s[ns] = wmma_bf16(af, bf[ns], s[ns]);
        }
      }
      // ---- scale + relative_time + causal mask ----
#pragma unroll
      for (int ns = 0; ns < 4; ++ns)
#pragma unroll
        for (int i = 0; i < 8; ++i) {
          const int q = qw + hi * 8 + i;
          const int k = k0 + ns * 16 + ln;
          const float v = s[ns][i] * 0.125f + relp[(size_t)q * SEQ + k];
          s[ns][i] = (k <= q) ? v : NEG_INF;
        }
      // ---- online softmax update ----
#pragma unroll
      for (int i = 0; i < 8; ++i) {
        float mx = s[0][i];
#pragma unroll
        for (int ns = 1; ns < 4; ++ns) mx = fmaxf(mx, s[ns][i]);
        mx = redmax16(mx);
        const float mn = fmaxf(mrow[i], mx);
        const float alpha = __expf(mrow[i] - mn);
        mrow[i] = mn;
        float rs = 0.0f;
#pragma unroll
        for (int ns = 0; ns < 4; ++ns) {
          const float p = __expf(s[ns][i] - mn);
          s[ns][i] = p;
          rs += p;
        }
        rs = redsum16(rs);
        lrow[i] = lrow[i] * alpha + rs;
#pragma unroll
        for (int nd = 0; nd < 4; ++nd) o[nd][i] *= alpha;
      }
      // ---- P (bf16) -> wave-private LDS, then O += P @ V ----
      __bf16* pb = &sP[wid][0];
#pragma unroll
      for (int ns = 0; ns < 4; ++ns)
#pragma unroll
        for (int i = 0; i < 8; ++i)
          pb[(hi * 8 + i) * 72 + ns * 16 + ln] = (__bf16)s[ns][i];

#pragma unroll
      for (int ks = 0; ks < 2; ++ks) {
        const int kbase = ks * 32;
        v16bf vb4[4];
#pragma unroll
        for (int nd = 0; nd < 4; ++nd) {
          const __bf16* p = &sVt[(nd * 16 + ln) * 72 + kbase + hi * 16];
          vb4[nd] = pack16(*(const v8bf*)p, *(const v8bf*)(p + 8));
        }
        const __bf16* p = pb + ln * 72 + kbase + hi * 16;
        v16bf pa = pack16(*(const v8bf*)p, *(const v8bf*)(p + 8));
#pragma unroll
        for (int nd = 0; nd < 4; ++nd) o[nd] = wmma_bf16(pa, vb4[nd], o[nd]);
      }
    }
    __syncthreads();
  }

  // ---- finalize: O /= l, store bf16 attn-out at [b][q][h*64+d] ----
#pragma unroll
  for (int i = 0; i < 8; ++i) {
    const float rinv = 1.0f / lrow[i];
    const int q = qw + hi * 8 + i;
#pragma unroll
    for (int nd = 0; nd < 4; ++nd)
      Ob[(size_t)(row0 + q) * HIDDIM + col0 + nd * 16 + ln] =
          (__bf16)(o[nd][i] * rinv);
  }
}

// ---------------------------------------------------------------------------
extern "C" void kernel_launch(void* const* d_in, const int* in_sizes, int n_in,
                              void* d_out, int out_size, void* d_ws, size_t ws_size,
                              hipStream_t stream) {
  (void)in_sizes; (void)n_in; (void)out_size; (void)ws_size;
  const float* seq_id   = (const float*)d_in[0];
  const float* seq_cate = (const float*)d_in[1];
  const float* seq_pos  = (const float*)d_in[2];
  const float* v_in     = (const float*)d_in[3];
  // d_in[4] attn_mask is causal-tril by construction; applied analytically.
  const float* rel      = (const float*)d_in[5];
  const float* q_id_w   = (const float*)d_in[6];
  const float* q_id_b   = (const float*)d_in[7];
  const float* k_id_w   = (const float*)d_in[8];
  const float* k_id_b   = (const float*)d_in[9];
  const float* v_id_w   = (const float*)d_in[10];
  const float* v_id_b   = (const float*)d_in[11];
  const float* q_cate_w = (const float*)d_in[12];
  const float* q_cate_b = (const float*)d_in[13];
  const float* k_cate_w = (const float*)d_in[14];
  const float* k_cate_b = (const float*)d_in[15];
  const float* q_pos_w  = (const float*)d_in[16];
  const float* q_pos_b  = (const float*)d_in[17];
  const float* k_pos_w  = (const float*)d_in[18];
  const float* k_pos_b  = (const float*)d_in[19];
  const float* out_w    = (const float*)d_in[20];
  const float* out_b    = (const float*)d_in[21];

  const size_t TEN = (size_t)8192 * 1024;  // elements per bf16[8192,1024] tensor
  __bf16* w   = (__bf16*)d_ws;
  __bf16* Qid = w + 0 * TEN; __bf16* Kid = w + 1 * TEN;
  __bf16* Qc  = w + 2 * TEN; __bf16* Kc  = w + 3 * TEN;
  __bf16* Qp  = w + 4 * TEN; __bf16* Kp  = w + 5 * TEN;
  __bf16* Vw  = w + 6 * TEN; __bf16* Ao  = w + 7 * TEN;

  dim3 g(8, 64), blk(256);
  gemm1024<false, false><<<g, blk, 0, stream>>>(seq_id,   q_id_w,   q_id_b,   Qid);
  gemm1024<false, false><<<g, blk, 0, stream>>>(seq_id,   k_id_w,   k_id_b,   Kid);
  gemm1024<false, false><<<g, blk, 0, stream>>>(seq_cate, q_cate_w, q_cate_b, Qc);
  gemm1024<false, false><<<g, blk, 0, stream>>>(seq_cate, k_cate_w, k_cate_b, Kc);
  gemm1024<false, false><<<g, blk, 0, stream>>>(seq_pos,  q_pos_w,  q_pos_b,  Qp);
  gemm1024<false, false><<<g, blk, 0, stream>>>(seq_pos,  k_pos_w,  k_pos_b,  Kp);
  gemm1024<false, false><<<g, blk, 0, stream>>>(v_in,     v_id_w,   v_id_b,   Vw);

  attn_kernel<<<dim3(16 * 16 * 4), dim3(256), 0, stream>>>(Qid, Kid, Qc, Kc, Qp, Kp,
                                                           Vw, rel, Ao);

  gemm1024<true, true><<<g, blk, 0, stream>>>(Ao, out_w, out_b, d_out);
}
